// Router_88081189306697
// MI455X (gfx1250) — matrix operand order repaired and checked by reading, not compiled, not run on hardware
//
#include <hip/hip_runtime.h>
#include <hip/hip_bf16.h>

// ---------------------------------------------------------------------------
// MoE router: logits = x @ W_gate^T, softmax over E=64, top-2, dispatch mask,
// normalized combine weights, indices.
// B=4, S=4096, D=4096, E=64, K=2  -> tokens M = 16384.
// bf16 WMMA GEMM (x converted on the fly, W pre-converted to d_ws once),
// f32 accumulate; per-token softmax/top-2 epilogue via LDS.
// ---------------------------------------------------------------------------

#define D_DIM 4096
#define E_DIM 64
#define TOKENS 16384          // B*S
#define TOPK 2

#define ROWS_PER_WAVE 32      // 2 M-tiles of 16
#define WAVES_PER_BLOCK 4
#define BLOCK_THREADS (WAVES_PER_BLOCK * 32)
#define NUM_BLOCKS (TOKENS / (ROWS_PER_WAVE * WAVES_PER_BLOCK))  // 128
#define LDS_STRIDE 66         // floats per row (64 + pad, conflict-free)

typedef __bf16 v16bf __attribute__((ext_vector_type(16)));
typedef __bf16 v4bf  __attribute__((ext_vector_type(4)));
typedef float  v8f   __attribute__((ext_vector_type(8)));
typedef float  v4f   __attribute__((ext_vector_type(4)));

// ---------------------------------------------------------------------------
// Kernel 1: convert W_gate [E,D] fp32 -> bf16 in workspace (512 KB).
// 65536 float4 elements, 256 blocks x 256 threads.
// ---------------------------------------------------------------------------
__global__ __launch_bounds__(256) void wcvt_kernel(const float4* __restrict__ W,
                                                   v4bf* __restrict__ Wb) {
  int i = blockIdx.x * blockDim.x + threadIdx.x;  // 0 .. 65535
  float4 f = W[i];
  v4bf o;
  o[0] = (__bf16)f.x;
  o[1] = (__bf16)f.y;
  o[2] = (__bf16)f.z;
  o[3] = (__bf16)f.w;
  Wb[i] = o;
}

// ---------------------------------------------------------------------------
// Kernel 2: GEMM (bf16 WMMA) + softmax + top-2 + outputs.
// ---------------------------------------------------------------------------
__global__ __launch_bounds__(BLOCK_THREADS) void router_kernel(
    const float* __restrict__ X,        // [TOKENS, D] fp32
    const __bf16* __restrict__ Wb,      // [E, D] bf16
    float* __restrict__ mask,           // [TOKENS, E]
    float* __restrict__ comb,           // [TOKENS, 2]
    float* __restrict__ idxf) {         // [TOKENS, 2] (indices as float)
  __shared__ float lds[WAVES_PER_BLOCK][ROWS_PER_WAVE][LDS_STRIDE];

  const int lane = threadIdx.x & 31;
  const int wave = threadIdx.x >> 5;
  const int l16 = lane & 15;
  const int khalf = lane >> 4;  // 0 or 1
  const int rowBase = (blockIdx.x * WAVES_PER_BLOCK + wave) * ROWS_PER_WAVE;

  // B-matrix base pointers: 4 N-tiles, expert e = nt*16 + l16.
  // B k order used here (not AMD's table order): lane half selects K 0..15 vs 16..31 (contiguous).
  const __bf16* bptr0 = Wb + (size_t)(0 * 16 + l16) * D_DIM + khalf * 16;
  const __bf16* bptr1 = Wb + (size_t)(1 * 16 + l16) * D_DIM + khalf * 16;
  const __bf16* bptr2 = Wb + (size_t)(2 * 16 + l16) * D_DIM + khalf * 16;
  const __bf16* bptr3 = Wb + (size_t)(3 * 16 + l16) * D_DIM + khalf * 16;

  // A-matrix base pointers: 2 M-tiles, row m = rowBase + mt*16 + l16.
  // A k order used here (not AMD's table order): lane half selects K 0..15 vs 16..31 (contiguous),
  // matching B.
  const float* aptr0 = X + (size_t)(rowBase + 0 * 16 + l16) * D_DIM + khalf * 16;
  const float* aptr1 = X + (size_t)(rowBase + 1 * 16 + l16) * D_DIM + khalf * 16;

  v8f acc[2][4] = {};

  for (int k0 = 0; k0 < D_DIM; k0 += 32) {
    v16bf bfrag0 = *(const v16bf*)(bptr0 + k0);
    v16bf bfrag1 = *(const v16bf*)(bptr1 + k0);
    v16bf bfrag2 = *(const v16bf*)(bptr2 + k0);
    v16bf bfrag3 = *(const v16bf*)(bptr3 + k0);

#pragma unroll
    for (int mt = 0; mt < 2; ++mt) {
      const float* ap = (mt == 0) ? aptr0 : aptr1;
      v4f a0 = *(const v4f*)(ap + k0);
      v4f a1 = *(const v4f*)(ap + k0 + 4);
      v4f a2 = *(const v4f*)(ap + k0 + 8);
      v4f a3 = *(const v4f*)(ap + k0 + 12);
      v16bf afrag;
#pragma unroll
      for (int j = 0; j < 4; ++j) {
        afrag[j]      = (__bf16)a0[j];
        afrag[4 + j]  = (__bf16)a1[j];
        afrag[8 + j]  = (__bf16)a2[j];
        afrag[12 + j] = (__bf16)a3[j];
      }
      acc[mt][0] = __builtin_amdgcn_wmma_f32_16x16x32_bf16(
          false, afrag, false, bfrag0, (short)0, acc[mt][0], false, false);
      acc[mt][1] = __builtin_amdgcn_wmma_f32_16x16x32_bf16(
          false, afrag, false, bfrag1, (short)0, acc[mt][1], false, false);
      acc[mt][2] = __builtin_amdgcn_wmma_f32_16x16x32_bf16(
          false, afrag, false, bfrag2, (short)0, acc[mt][2], false, false);
      acc[mt][3] = __builtin_amdgcn_wmma_f32_16x16x32_bf16(
          false, afrag, false, bfrag3, (short)0, acc[mt][3], false, false);
    }
  }

  // Stage logits to LDS. C/D layout: lane holds column n = l16 (per N-tile),
  // VGPR r holds row m = r + 8*khalf.
  float(*myLds)[LDS_STRIDE] = lds[wave];
#pragma unroll
  for (int mt = 0; mt < 2; ++mt) {
#pragma unroll
    for (int nt = 0; nt < 4; ++nt) {
#pragma unroll
      for (int r = 0; r < 8; ++r) {
        int row = mt * 16 + khalf * 8 + r;
        int col = nt * 16 + l16;
        myLds[row][col] = acc[mt][nt][r];
      }
    }
  }
  // Same-wave LDS RAW: DScnt wait inserted by compiler; no barrier needed.

  // Each lane owns one token row (32 rows per wave, 32 lanes).
  const float* lr = myLds[lane];

  // Pass 1: top-2 over logits (softmax is monotonic; ties -> lowest index,
  // matching jax.lax.top_k).
  float v0 = -3.0e38f, v1 = -3.0e38f;
  int i0 = 0, i1 = 0;
#pragma unroll 4
  for (int j = 0; j < E_DIM; ++j) {
    float v = lr[j];
    if (v > v0) {
      v1 = v0; i1 = i0;
      v0 = v;  i0 = j;
    } else if (v > v1) {
      v1 = v; i1 = j;
    }
  }
  // Pass 2: softmax denominator with max subtraction.
  float sum = 0.0f;
#pragma unroll 4
  for (int j = 0; j < E_DIM; ++j) sum += __expf(lr[j] - v0);
  float inv = 1.0f / sum;
  float s0 = __expf(v0 - v0) * inv;  // = inv
  float s1 = __expf(v1 - v0) * inv;
  float denom = s0 + s1 + 1e-9f;
  float w0 = s0 / denom;
  float w1 = s1 / denom;

  size_t t = (size_t)rowBase + lane;

  // Dispatch mask: 64 floats, one-hot sum at i0, i1.
  float* mrow = mask + t * E_DIM;
#pragma unroll
  for (int j4 = 0; j4 < 16; ++j4) {
    v4f m;
#pragma unroll
    for (int c = 0; c < 4; ++c) {
      int j = j4 * 4 + c;
      m[c] = (j == i0 || j == i1) ? 1.0f : 0.0f;
    }
    *(v4f*)(mrow + j4 * 4) = m;
  }

  comb[t * 2 + 0] = w0;
  comb[t * 2 + 1] = w1;
  idxf[t * 2 + 0] = (float)i0;
  idxf[t * 2 + 1] = (float)i1;
}

// ---------------------------------------------------------------------------
extern "C" void kernel_launch(void* const* d_in, const int* in_sizes, int n_in,
                              void* d_out, int out_size, void* d_ws, size_t ws_size,
                              hipStream_t stream) {
  (void)in_sizes; (void)n_in; (void)out_size; (void)ws_size;

  const float* x = (const float*)d_in[0];       // [B,S,D] fp32
  const float* wg = (const float*)d_in[1];      // [E,D] fp32

  __bf16* wb = (__bf16*)d_ws;                   // [E,D] bf16 (512 KB)

  float* mask = (float*)d_out;                          // [TOKENS, E]
  float* comb = mask + (size_t)TOKENS * E_DIM;          // [TOKENS, 2]
  float* idxf = comb + (size_t)TOKENS * TOPK;           // [TOKENS, 2]

  // W_gate fp32 -> bf16 (E*D/4 = 65536 float4s).
  wcvt_kernel<<<(E_DIM * D_DIM / 4) / 256, 256, 0, stream>>>(
      (const float4*)wg, (v4bf*)wb);

  router_kernel<<<NUM_BLOCKS, BLOCK_THREADS, 0, stream>>>(x, wb, mask, comb,
                                                          idxf);
}
